// epipolar_Attention_Block_67413806678640
// MI455X (gfx1250) — compile-verified, matrix operand order checked by reading, not run on hardware
//
#include <hip/hip_runtime.h>
#include <hip/hip_bf16.h>

typedef __attribute__((ext_vector_type(16))) _Float16 v16h;
typedef __attribute__((ext_vector_type(8)))  float    v8f;

#define CH  128
#define HW  1024
#define RES 32

// ---------------------------------------------------------------------------
// Kernel 1: bilinear 2x downsample (== 2x2 average pool with half-pixel centers)
// key (128,64,64) -> kds (128,32,32)
// ---------------------------------------------------------------------------
__global__ __launch_bounds__(256) void k_downsample(const float* __restrict__ key,
                                                    float* __restrict__ kds) {
    int idx = blockIdx.x * blockDim.x + threadIdx.x;   // 0 .. CH*HW-1
    if (idx >= CH * HW) return;
    int c = idx >> 10;
    int p = idx & 1023;
    int y = p >> 5, x = p & 31;
    const float* kc = key + (size_t)c * 4096;
    int r0 = (2 * y) * 64 + 2 * x;
    kds[idx] = 0.25f * (kc[r0] + kc[r0 + 1] + kc[r0 + 64] + kc[r0 + 65]);
}

// ---------------------------------------------------------------------------
// Kernel 2: per-channel masked-affinity softmax-weighted sum via WMMA.
// Grid: CH*16 blocks of 128 threads (4 waves); each wave owns a 16-row block.
//   attn[c,i] = sum_j e_ij * k_j / sum_j e_ij,  e_ij = exp(k_i*q_j*W_ij - m_i)
// Both j-reductions are one 16xK x Kx16 WMMA: B col0 = ones, col1 = k_j.
// ---------------------------------------------------------------------------
__global__ __launch_bounds__(128) void k_attn(const float* __restrict__ kds,
                                              const float* __restrict__ q,
                                              const float* __restrict__ W,
                                              float* __restrict__ attn) {
    __shared__ alignas(16) float lq[HW];
    __shared__ alignas(16) float lk[HW];
    __shared__ float red[4][32];

    int c   = blockIdx.x >> 4;         // channel
    int rb4 = blockIdx.x & 15;         // group of 4 row-blocks
    int tid = threadIdx.x;

    // stage q, k for this channel (shared by the 4 waves), vectorized
    {
        const float4* qg = (const float4*)(q   + (size_t)c * HW);
        const float4* kg = (const float4*)(kds + (size_t)c * HW);
        float4* ql = (float4*)lq;
        float4* kl = (float4*)lk;
        for (int i = tid; i < HW / 4; i += 128) {
            ql[i] = qg[i];
            kl[i] = kg[i];
        }
    }
    __syncthreads();

    int wave  = tid >> 5;
    int lane  = tid & 31;
    int rb    = rb4 * 4 + wave;        // row block 0..63
    int i0    = rb * 16;
    int M     = lane & 15;             // row within tile (A) / column N (B,D)
    int khalf = lane >> 4;             // which K-half this lane covers
    const float* wrow = W + (size_t)(i0 + M) * HW;
    float ki = lk[i0 + M];

    // ---- Pass A: row max of s_ij = ki * (q_j * W_ij) --------------------
    // lane pair (L, L+16) split the 1024 columns of row M; float4 streams.
    float mx = -1e30f, mn = 1e30f;
    int jbeg = khalf * 512;
    for (int j = jbeg; j < jbeg + 512; j += 4) {
        float4 wv = *(const float4*)(wrow + j);
        float4 qv = *(const float4*)(lq + j);
        float p0 = qv.x * wv.x, p1 = qv.y * wv.y;
        float p2 = qv.z * wv.z, p3 = qv.w * wv.w;
        mx = fmaxf(mx, fmaxf(fmaxf(p0, p1), fmaxf(p2, p3)));
        mn = fminf(mn, fminf(fminf(p0, p1), fminf(p2, p3)));
    }
    mx = fmaxf(mx, __shfl_xor(mx, 16, 32));
    mn = fminf(mn, __shfl_xor(mn, 16, 32));
    float nm = -((ki >= 0.f) ? ki * mx : ki * mn);   // -(exact row max of s)

    // loop-invariant per-lane column selectors for B
    float sel0 = (M == 0) ? 1.0f : 0.0f;   // denominator column (ones)
    float sel1 = (M == 1) ? 1.0f : 0.0f;   // numerator column (k_j)

    // ---- Pass B: exp + WMMA accumulation over 32-column chunks ----------
    v8f acc = {};
#pragma unroll 2
    for (int chunk = 0; chunk < 32; ++chunk) {
        int j0  = chunk * 32;
        __builtin_prefetch(wrow + j0 + 128, 0, 1);   // pull upcoming W lines
        // lane covers two contiguous 8-float runs of K:
        //   run0: K = 8*khalf + [0..7]   -> a[e], e = 0..7
        //   run1: K = 16 + 8*khalf + [0..7] -> a[e], e = 8..15
        int jb0 = j0 + 8 * khalf;
        int jb1 = j0 + 16 + 8 * khalf;

        float w[16], qq[16], kk[16];
        *(float4*)(&w[0])   = *(const float4*)(wrow + jb0);
        *(float4*)(&w[4])   = *(const float4*)(wrow + jb0 + 4);
        *(float4*)(&w[8])   = *(const float4*)(wrow + jb1);
        *(float4*)(&w[12])  = *(const float4*)(wrow + jb1 + 4);
        *(float4*)(&qq[0])  = *(const float4*)(lq + jb0);
        *(float4*)(&qq[4])  = *(const float4*)(lq + jb0 + 4);
        *(float4*)(&qq[8])  = *(const float4*)(lq + jb1);
        *(float4*)(&qq[12]) = *(const float4*)(lq + jb1 + 4);
        *(float4*)(&kk[0])  = *(const float4*)(lk + jb0);
        *(float4*)(&kk[4])  = *(const float4*)(lk + jb0 + 4);
        *(float4*)(&kk[8])  = *(const float4*)(lk + jb1);
        *(float4*)(&kk[12]) = *(const float4*)(lk + jb1 + 4);

        v16h a, b;
#pragma unroll
        for (int e = 0; e < 16; ++e) {
            float t  = qq[e] * w[e];
            float ev = __expf(fmaf(ki, t, nm));        // in (0,1], f16-safe
            a[e] = (_Float16)ev;
            // branch-free column select: N==0 -> 1.0, N==1 -> k_j, else 0
            b[e] = (_Float16)fmaf(sel1, kk[e], sel0);
        }
        acc = __builtin_amdgcn_wmma_f32_16x16x32_f16(
                  /*neg_a=*/false, a, /*neg_b=*/false, b,
                  /*c_mod=*/(short)0, acc, /*reuse_a=*/false, /*reuse_b=*/false);
    }

    // ---- Extract den (D[:,0]) and num (D[:,1]) per f32 C/D layout -------
    // rows 0-7 live in lanes 0/1 (vgpr v = row), rows 8-15 in lanes 16/17.
    if (lane == 0 || lane == 1 || lane == 16 || lane == 17) {
        int colsel = lane & 15;            // 0 = den, 1 = num
        int rowoff = (lane >> 4) * 8;
#pragma unroll
        for (int v = 0; v < 8; ++v)
            red[wave][colsel * 16 + rowoff + v] = acc[v];
    }
    __syncthreads();
    if (lane < 16) {
        float den = red[wave][lane];
        float num = red[wave][16 + lane];
        attn[(size_t)c * HW + i0 + lane] = num / den;
    }
}

// ---------------------------------------------------------------------------
// Kernel 3: per-channel softmax over HW + residual add of query.
// ---------------------------------------------------------------------------
__global__ __launch_bounds__(256) void k_softmax_add(const float* __restrict__ attn,
                                                     const float* __restrict__ q,
                                                     float* __restrict__ out) {
    __shared__ float sred[256];
    int c   = blockIdx.x;
    int tid = threadIdx.x;
    const float* a = attn + (size_t)c * HW;

    float mx = -1e30f;
    for (int i = tid; i < HW; i += 256) mx = fmaxf(mx, a[i]);
    sred[tid] = mx;
    __syncthreads();
    for (int s = 128; s > 0; s >>= 1) {
        if (tid < s) sred[tid] = fmaxf(sred[tid], sred[tid + s]);
        __syncthreads();
    }
    mx = sred[0];
    __syncthreads();

    float sum = 0.f;
    for (int i = tid; i < HW; i += 256) sum += __expf(a[i] - mx);
    sred[tid] = sum;
    __syncthreads();
    for (int s = 128; s > 0; s >>= 1) {
        if (tid < s) sred[tid] += sred[tid + s];
        __syncthreads();
    }
    float inv = 1.f / sred[0];

    for (int i = tid; i < HW; i += 256)
        out[(size_t)c * HW + i] = __expf(a[i] - mx) * inv + q[(size_t)c * HW + i];
}

// ---------------------------------------------------------------------------
extern "C" void kernel_launch(void* const* d_in, const int* in_sizes, int n_in,
                              void* d_out, int out_size, void* d_ws, size_t ws_size,
                              hipStream_t stream) {
    const float* key   = (const float*)d_in[0];   // (1,128,64,64)
    const float* query = (const float*)d_in[1];   // (1,128,32,32)
    // d_in[2] = t, unused by the reference
    const float* W     = (const float*)d_in[3];   // (1024,1024)
    float* out = (float*)d_out;                    // (1,128,32,32)

    float* kds  = (float*)d_ws;                    // 128*1024 floats (512 KB)
    float* attn = kds + (size_t)CH * HW;           // 128*1024 floats (512 KB)

    k_downsample<<<(CH * HW + 255) / 256, 256, 0, stream>>>(key, kds);
    k_attn<<<CH * 16, 128, 0, stream>>>(kds, query, W, attn);
    k_softmax_add<<<CH, 256, 0, stream>>>(attn, query, out);
}
